// Attention_60825326846013
// MI455X (gfx1250) — matrix operand order repaired and checked by reading, not compiled, not run on hardware
//
#include <hip/hip_runtime.h>
#include <hip/hip_bf16.h>
#include <math.h>

typedef __bf16 bf16;
typedef __attribute__((ext_vector_type(16))) __bf16 v16bf;
typedef __attribute__((ext_vector_type(8)))  __bf16 v8bf;
typedef __attribute__((ext_vector_type(4)))  __bf16 v4bf;
typedef __attribute__((ext_vector_type(8)))  float   v8f;

#define B_  2
#define N_  2048
#define C_  1024
#define H_  16
#define D_  64
#define SCALE_ 0.125f   // 64^-0.5

// ---------------------------------------------------------------- helpers

__device__ __forceinline__ v8f wmma_bf16(v16bf a, v16bf b, v8f c) {
  // D = A(16x32 bf16) x B(32x16 bf16) + C(f32)
  return __builtin_amdgcn_wmma_f32_16x16x32_bf16(
      /*neg_a=*/false, a, /*neg_b=*/false, b,
      /*c_mod=*/(short)0, c, /*reuse_a=*/false, /*reuse_b=*/false);
}

// A fragment (16x32, row-major source, row stride = ld elements):
// lane l: row m=l&15; elems 0..15 = K kc+16h+0..15
// caller passes p = &src[m*ld + kc + 16*half]
__device__ __forceinline__ v16bf load_fragA(const bf16* p) {
  union { v16bf v; v8bf h[2]; } u;
  u.h[0] = *(const v8bf*)p;
  u.h[1] = *(const v8bf*)(p + 8);
  return u.v;
}

// B fragment (32x16): lane l: col n=l&15; elems 0..15 = K kc+16*half+0..15
// caller passes p = &srcT[n*ld + kc + 16*half]   (srcT row-major along K)
__device__ __forceinline__ v16bf load_fragB(const bf16* p) {
  return *(const v16bf*)p;
}

__device__ __forceinline__ v8f vzero8() {
  v8f z;
  #pragma unroll
  for (int i = 0; i < 8; i++) z[i] = 0.0f;
  return z;
}

// ---------------------------------------------------------------- fp32 -> bf16

__global__ void cvt_f32_to_bf16(const float* __restrict__ s,
                                bf16* __restrict__ d, int n4) {
  int i = blockIdx.x * blockDim.x + threadIdx.x;
  int stride = gridDim.x * blockDim.x;
  for (; i < n4; i += stride) {
    float4 f = ((const float4*)s)[i];
    v4bf o;
    o[0] = (bf16)f.x; o[1] = (bf16)f.y; o[2] = (bf16)f.z; o[3] = (bf16)f.w;
    ((v4bf*)d)[i] = o;
  }
}

// ---------------------------------------------------------------- QKV GEMM
// y = x @ W.T for W rows [Wq(1024) | Wkv(2048)], fused. Wave: 16 rows x 64 cols.
// Q,K stored [b][h][n][d] bf16; V stored transposed [b][h][d][n] bf16.

__global__ __launch_bounds__(128) void qkv_gemm(
    const bf16* __restrict__ xb, const bf16* __restrict__ wqb,
    const bf16* __restrict__ wkvb,
    bf16* __restrict__ qo, bf16* __restrict__ ko, bf16* __restrict__ vto) {
  const int lane = threadIdx.x & 31;
  const int wid  = blockIdx.x * (blockDim.x >> 5) + (threadIdx.x >> 5);
  const int rowTile = wid / 48;            // 256 tiles of 16 rows
  const int colTile = wid % 48;            // 48 tiles of 64 cols (3072 total)
  const int l15 = lane & 15, hf = lane >> 4;
  const int off8 = hf * 8, off16 = hf * 16;
  const int m0 = rowTile * 16, c0 = colTile * 64;
  const int mrow = m0 + l15;

  const bf16* wb; int w0;
  if (c0 < C_) { wb = wqb;  w0 = c0;      }
  else         { wb = wkvb; w0 = c0 - C_; }

  v8f acc[4];
  #pragma unroll
  for (int s = 0; s < 4; s++) acc[s] = vzero8();

  const bf16* arow = xb + (size_t)mrow * C_;
  for (int kc = 0; kc < C_; kc += 32) {
    v16bf a = load_fragA(arow + kc + off16);
    #pragma unroll
    for (int s = 0; s < 4; s++) {
      v16bf bfr = load_fragB(wb + (size_t)(w0 + s * 16 + l15) * C_ + kc + off16);
      acc[s] = wmma_bf16(a, bfr, acc[s]);
    }
  }

  // scatter-store: C layout lane l -> col = c0+16s+(l&15); vgpr r -> row m0+r+8*hf
  #pragma unroll
  for (int s = 0; s < 4; s++) {
    const int col = c0 + s * 16 + l15;
    #pragma unroll
    for (int r = 0; r < 8; r++) {
      const int row = m0 + r + off8;
      const int bb = row >> 11, n = row & (N_ - 1);
      bf16 v = (bf16)acc[s][r];
      if (col < C_) {                       // Q
        const int hh = col >> 6, d = col & 63;
        qo[(((size_t)bb * H_ + hh) * N_ + n) * D_ + d] = v;
      } else if (col < 2 * C_) {            // K
        const int cc = col - C_, hh = cc >> 6, d = cc & 63;
        ko[(((size_t)bb * H_ + hh) * N_ + n) * D_ + d] = v;
      } else {                              // V transposed
        const int cc = col - 2 * C_, hh = cc >> 6, d = cc & 63;
        vto[(((size_t)bb * H_ + hh) * D_ + d) * N_ + n] = v;
      }
    }
  }
}

// ---------------------------------------------------------------- flash attention
// Per wave: one (b,h) and 16 query rows; iterate keys in blocks of 32.
// S^T = K . Q^T  (softmax dim along VGPRs),  O^T = V^T . P^T. LDS-free.

__global__ __launch_bounds__(128) void attn_kernel(
    const bf16* __restrict__ qb, const bf16* __restrict__ kbuf,
    const bf16* __restrict__ vtb, const float* __restrict__ bias,
    bf16* __restrict__ aout) {
  const int lane = threadIdx.x & 31;
  const int wid  = blockIdx.x * (blockDim.x >> 5) + (threadIdx.x >> 5); // 0..4095
  const int qtile = wid & 127;    // N/16
  const int bh    = wid >> 7;     // b*H + h, 0..31
  const int b     = bh >> 4;
  const int h     = bh & 15;
  const int l15 = lane & 15, hf = lane >> 4;
  const int off8 = hf * 8, off16 = hf * 16;
  const int nq = qtile * 16 + l15;

  const bf16*  qrow   = qb   + ((size_t)bh * N_ + nq) * D_;
  const bf16*  khead  = kbuf + (size_t)bh * N_ * D_;
  const bf16*  vthead = vtb  + (size_t)bh * D_ * N_;
  const float* brow   = bias + ((size_t)bh * N_ + nq) * (size_t)N_;

  const v16bf qf0 = load_fragB(qrow + off16);        // d 0..31
  const v16bf qf1 = load_fragB(qrow + 32 + off16);   // d 32..63

  v8f oacc[4];
  #pragma unroll
  for (int t = 0; t < 4; t++) oacc[t] = vzero8();
  float mrun = -INFINITY;
  float lrun = 0.0f;

  for (int kb32 = 0; kb32 < N_; kb32 += 32) {
    // S^T tiles: tile0 = keys kb32..+15, tile1 = keys kb32+16..+31
    const bf16* krow0 = khead + (size_t)(kb32 + l15) * D_;
    const bf16* krow1 = krow0 + 16 * D_;
    v8f s0 = wmma_bf16(load_fragA(krow0 + off16),      qf0, vzero8());
    s0     = wmma_bf16(load_fragA(krow0 + 32 + off16), qf1, s0);
    v8f s1 = wmma_bf16(load_fragA(krow1 + off16),      qf0, vzero8());
    s1     = wmma_bf16(load_fragA(krow1 + 32 + off16), qf1, s1);

    // scale + bias. C layout: vgpr r -> key kb32 + r + 8*hf (tile0), +16 (tile1)
    float sv0[8], sv1[8];
    {
      const float* bp0 = brow + kb32 + off8;
      const float* bp1 = brow + kb32 + 16 + off8;
      float4 a0 = *(const float4*)(bp0);
      float4 a1 = *(const float4*)(bp0 + 4);
      float4 b0 = *(const float4*)(bp1);
      float4 b1 = *(const float4*)(bp1 + 4);
      float bb0[8] = {a0.x, a0.y, a0.z, a0.w, a1.x, a1.y, a1.z, a1.w};
      float bb1[8] = {b0.x, b0.y, b0.z, b0.w, b1.x, b1.y, b1.z, b1.w};
      #pragma unroll
      for (int r = 0; r < 8; r++) {
        sv0[r] = fmaf(s0[r], SCALE_, bb0[r]);
        sv1[r] = fmaf(s1[r], SCALE_, bb1[r]);
      }
    }

    // online softmax (lane pair (l, l+16) share query l&15 -> sync max)
    float mloc = sv0[0];
    #pragma unroll
    for (int r = 0; r < 8; r++) {
      mloc = fmaxf(mloc, sv0[r]);
      mloc = fmaxf(mloc, sv1[r]);
    }
    mloc = fmaxf(mloc, __shfl_xor(mloc, 16, 32));
    const float mnew = fmaxf(mrun, mloc);
    const float rescale = __expf(mrun - mnew);
    mrun = mnew;
    lrun *= rescale;
    #pragma unroll
    for (int t = 0; t < 4; t++)
      #pragma unroll
      for (int r = 0; r < 8; r++) oacc[t][r] *= rescale;

    // p = exp(s - m); pack to bf16 pairs
    unsigned p0pk[4], p1pk[4];
    float psum = 0.0f;
    #pragma unroll
    for (int rp = 0; rp < 4; rp++) {
      float e0a = __expf(sv0[2 * rp]     - mnew);
      float e0b = __expf(sv0[2 * rp + 1] - mnew);
      float e1a = __expf(sv1[2 * rp]     - mnew);
      float e1b = __expf(sv1[2 * rp + 1] - mnew);
      psum += (e0a + e0b) + (e1a + e1b);
      union { bf16 b[2]; unsigned u; } t0, t1;
      t0.b[0] = (bf16)e0a; t0.b[1] = (bf16)e0b; p0pk[rp] = t0.u;
      t1.b[0] = (bf16)e1a; t1.b[1] = (bf16)e1b; p1pk[rp] = t1.u;
    }
    lrun += psum;

    // Assemble P^T B-fragment: lane half 0 needs keys kb32+0..15, half 1 needs +16..31.
    // own tile0 covers keys +8*hf..+8*hf+7; exchange the other half with partner lane.
    union { v16bf v; unsigned u[8]; } pf;
    #pragma unroll
    for (int rp = 0; rp < 4; rp++) {
      unsigned send = hf ? p0pk[rp] : p1pk[rp];
      unsigned recv = (unsigned)__shfl_xor((int)send, 16, 32);
      pf.u[rp]     = hf ? recv     : p0pk[rp];
      pf.u[4 + rp] = hf ? p1pk[rp] : recv;
    }

    // O^T += V^T . P^T   (4 d-tiles of 16)
    const bf16* vt0 = vthead + (size_t)l15 * N_ + kb32 + off16;
    oacc[0] = wmma_bf16(load_fragA(vt0),            pf.v, oacc[0]);
    oacc[1] = wmma_bf16(load_fragA(vt0 + 16 * N_),  pf.v, oacc[1]);
    oacc[2] = wmma_bf16(load_fragA(vt0 + 32 * N_),  pf.v, oacc[2]);
    oacc[3] = wmma_bf16(load_fragA(vt0 + 48 * N_),  pf.v, oacc[3]);
  }

  const float ltot = lrun + __shfl_xor(lrun, 16, 32);
  const float rinv = 1.0f / ltot;

  // O^T C layout: lane -> query nq; vgpr r -> d = 16t + r + 8*hf  (8 contiguous d)
  bf16* orow = aout + ((size_t)b * N_ + nq) * C_ + h * D_;
  #pragma unroll
  for (int t = 0; t < 4; t++) {
    v8bf pk;
    #pragma unroll
    for (int r = 0; r < 8; r++) pk[r] = (bf16)(oacc[t][r] * rinv);
    *(v8bf*)(orow + t * 16 + off8) = pk;
  }
}

// ---------------------------------------------------------------- output proj
// out = aout @ Wo.T + bo   (fp32 output)

__global__ __launch_bounds__(128) void oproj_gemm(
    const bf16* __restrict__ aout, const bf16* __restrict__ wob,
    const float* __restrict__ bo, float* __restrict__ out) {
  const int lane = threadIdx.x & 31;
  const int wid  = blockIdx.x * (blockDim.x >> 5) + (threadIdx.x >> 5); // 0..4095
  const int rowTile = wid >> 4;   // 256
  const int colTile = wid & 15;   // 16 tiles of 64 cols
  const int l15 = lane & 15, hf = lane >> 4;
  const int off8 = hf * 8, off16 = hf * 16;
  const int m0 = rowTile * 16, c0 = colTile * 64;
  const int mrow = m0 + l15;

  v8f acc[4];
  #pragma unroll
  for (int s = 0; s < 4; s++) acc[s] = vzero8();

  const bf16* arow = aout + (size_t)mrow * C_;
  for (int kc = 0; kc < C_; kc += 32) {
    v16bf a = load_fragA(arow + kc + off16);
    #pragma unroll
    for (int s = 0; s < 4; s++) {
      v16bf bfr = load_fragB(wob + (size_t)(c0 + s * 16 + l15) * C_ + kc + off16);
      acc[s] = wmma_bf16(a, bfr, acc[s]);
    }
  }

  #pragma unroll
  for (int s = 0; s < 4; s++) {
    const int col = c0 + s * 16 + l15;
    const float bv = bo[col];
    #pragma unroll
    for (int r = 0; r < 8; r++) {
      const int row = m0 + r + off8;
      out[(size_t)row * C_ + col] = acc[s][r] + bv;
    }
  }
}

// ---------------------------------------------------------------- launch

extern "C" void kernel_launch(void* const* d_in, const int* in_sizes, int n_in,
                              void* d_out, int out_size, void* d_ws, size_t ws_size,
                              hipStream_t stream) {
  const float* x    = (const float*)d_in[0];
  const float* bias = (const float*)d_in[1];
  const float* Wq   = (const float*)d_in[2];
  const float* Wkv  = (const float*)d_in[3];
  const float* Wo   = (const float*)d_in[4];
  const float* bo   = (const float*)d_in[5];
  float* out = (float*)d_out;

  char* p = (char*)d_ws;
  bf16* xb   = (bf16*)p; p += (size_t)(B_ * N_) * C_ * sizeof(bf16);   // 8 MB
  bf16* wqb  = (bf16*)p; p += (size_t)C_ * C_ * sizeof(bf16);          // 2 MB
  bf16* wkvb = (bf16*)p; p += (size_t)2 * C_ * C_ * sizeof(bf16);      // 4 MB
  bf16* wob  = (bf16*)p; p += (size_t)C_ * C_ * sizeof(bf16);          // 2 MB
  bf16* qb   = (bf16*)p; p += (size_t)(B_ * N_) * C_ * sizeof(bf16);   // 8 MB
  bf16* kb   = (bf16*)p; p += (size_t)(B_ * N_) * C_ * sizeof(bf16);   // 8 MB
  bf16* vtb  = (bf16*)p; p += (size_t)(B_ * N_) * C_ * sizeof(bf16);   // 8 MB
  bf16* aob  = (bf16*)p;                                               // 8 MB

  cvt_f32_to_bf16<<<1024, 256, 0, stream>>>(x,   xb,   (B_ * N_ * C_) / 4);
  cvt_f32_to_bf16<<<512,  256, 0, stream>>>(Wq,  wqb,  (C_ * C_) / 4);
  cvt_f32_to_bf16<<<512,  256, 0, stream>>>(Wkv, wkvb, (2 * C_ * C_) / 4);
  cvt_f32_to_bf16<<<512,  256, 0, stream>>>(Wo,  wob,  (C_ * C_) / 4);

  // 256 row-tiles x 48 col-tiles = 12288 waves, 4 waves/block
  qkv_gemm<<<3072, 128, 0, stream>>>(xb, wqb, wkvb, qb, kb, vtb);
  // B*H*(N/16) = 4096 waves
  attn_kernel<<<1024, 128, 0, stream>>>(qb, kb, vtb, bias, aob);
  // 256 x 16 = 4096 waves
  oproj_gemm<<<1024, 128, 0, stream>>>(aob, wob, bo, out);
}
